// PointFormer_32856499814497
// MI455X (gfx1250) — matrix operand order repaired and checked by reading, not compiled, not run on hardware
//
#include <hip/hip_runtime.h>
#include <hip/hip_bf16.h>
#include <stdint.h>

// ---------------------------------------------------------------------------
// PointFormer inference pipeline for gfx1250 (MI455X), bf16 WMMA GEMM core.
// B=16, N=1024, k=20. Dense GEMMs (Gram / GAB) use v_wmma_f32_16x16x32_bf16.
// Round 3 fix: async-copy builtin takes (v4i AS1*, v4i AS3*, imm, imm) where
// v4i = int __attribute__((vector_size(16))) -- cast accordingly (non-const).
// ---------------------------------------------------------------------------

typedef __attribute__((ext_vector_type(16))) __bf16 v16bf;
typedef __attribute__((ext_vector_type(8)))  float  v8f;
typedef int v4i __attribute__((vector_size(16)));

#define NPTS 1024
#define NB   16
#define KNN  20

#if __has_builtin(__builtin_amdgcn_global_load_async_to_lds_b128) && \
    __has_builtin(__builtin_amdgcn_s_wait_asynccnt)
#define HAVE_ASYNC_LDS 1
#else
#define HAVE_ASYNC_LDS 0
#endif

__device__ __forceinline__ float lrelu(float v) { return v > 0.f ? v : 0.2f * v; }

__device__ __forceinline__ unsigned short f2bf(float f) {
    unsigned int u = __float_as_uint(f);
    unsigned int r = (u + 0x7FFFu + ((u >> 16) & 1u)) >> 16;   // round-to-nearest-even
    return (unsigned short)r;
}
__device__ __forceinline__ float bf2f(unsigned short h) {
    return __uint_as_float(((unsigned int)h) << 16);
}

// ---------------------------------------------------------------------------
// Generic conversion kernels
// ---------------------------------------------------------------------------
__global__ void k_tobf(const float* __restrict__ in, unsigned short* __restrict__ out, long total) {
    long i = (long)blockIdx.x * blockDim.x + threadIdx.x;
    if (i < total) out[i] = f2bf(in[i]);
}

// x [B,C,N] (f32) -> out [B,N,C] (bf16)   (both operands of per-batch Gram GEMM)
__global__ void k_chw2nc(const float* __restrict__ x, unsigned short* __restrict__ out, int C, int N) {
    long i = (long)blockIdx.x * blockDim.x + threadIdx.x;
    long total = (long)NB * C * N;
    if (i >= total) return;
    long c = i % C;
    long n = (i / C) % N;
    long b = i / ((long)C * N);
    out[i] = f2bf(x[(b * C + c) * N + n]);
}

// xx[b,n] = sum_c x[b,c,n]^2
__global__ void k_xx(const float* __restrict__ x, float* __restrict__ xx, int C, int N) {
    int i = blockIdx.x * blockDim.x + threadIdx.x;
    if (i >= NB * N) return;
    int b = i / N, n = i % N;
    float a = 0.f;
    for (int c = 0; c < C; ++c) {
        float v = x[((long)b * C + c) * N + n];
        a += v * v;
    }
    xx[i] = a;
}

// x_glo[b, 0..N-1] = sum_c x ; x_glo[b, N..2N-1] = mean_c x   (C==3)
__global__ void k_xglo(const float* __restrict__ x, float* __restrict__ xg, int N) {
    int i = blockIdx.x * blockDim.x + threadIdx.x;
    if (i >= NB * N) return;
    int b = i / N, n = i % N;
    float s = x[((long)b * 3 + 0) * N + n] + x[((long)b * 3 + 1) * N + n] + x[((long)b * 3 + 2) * N + n];
    xg[(long)b * 2 * N + n]     = s;
    xg[(long)b * 2 * N + N + n] = s * (1.f / 3.f);
}

// ---------------------------------------------------------------------------
// Tiled bf16 GEMM with fp32 WMMA accumulation.
//   C[M,N] = A[M,K] * Bt[N,K]^T    (A, Bt bf16 row-major; per-batch strides)
// Epilogue flags: 1 = scale/bias (per column) + LeakyReLU
//                 2 = store bf16 (else fp32)
//                 4 = add rowbias[(row/rows_per_batch)*rb_ld + col] before scale
//
// LDS layouts are fragment-native (frag = two ds_load_b128):
//   As  [row][k]  row stride 40 u16 (80 B)
//   BsT [col][k]  col stride 40 u16 (80 B)
// ---------------------------------------------------------------------------
#define TM 64
#define TN 64
#define TK 32
#define LPAD 8   // pad to keep 16B alignment of fragment chunks

union FragU { uint4 q[2]; unsigned int u[8]; v16bf v; };

__global__ __launch_bounds__(128) void k_gemm_bf16(
    const unsigned short* __restrict__ A, const unsigned short* __restrict__ Bt,
    float* __restrict__ Cout, unsigned short* __restrict__ Cbf,
    int M, int N, int K,
    long sAb, long sBb, long sCb,
    const float* __restrict__ scale, const float* __restrict__ bias,
    const float* __restrict__ rowbias, int rows_per_batch, int rb_ld,
    int flags)
{
    __shared__ unsigned short As [TM][TK + LPAD];
    __shared__ unsigned short BsT[TN][TK + LPAD];

    int bz = blockIdx.z;
    A  += (long)bz * sAb;
    Bt += (long)bz * sBb;
    long cOff = (long)bz * sCb;

    int m0 = blockIdx.y * TM;
    int n0 = blockIdx.x * TN;
    int tid  = threadIdx.x;
    int wave = tid >> 5;
    int lane = tid & 31;
    int wr = (wave >> 1) * 32;   // wave row offset in block tile
    int wc = (wave & 1)  * 32;   // wave col offset

    v8f acc[2][2] = {};

    int mA    = lane & 15;
    int kSelA = (lane < 16) ? 0 : 8;    // A layout: hi lanes own K+8 / K+24
    int nBl   = lane & 15;
    int kSelB = (lane < 16) ? 0 : 8;    // B layout: hi lanes own K+8 / K+24

    const bool fast = ((K & 31) == 0) && ((M & 63) == 0) && ((N & 63) == 0);
    const int  sr = tid >> 1;            // 0..63: staged row (A) / col (Bt)
    const int  sh = (tid & 1) << 4;      // 0 or 16 (u16 units)

    for (int k0 = 0; k0 < K; k0 += TK) {
        if (fast) {
            const unsigned short* ga = &A [(long)(m0 + sr) * K + k0 + sh];
            const unsigned short* gb = &Bt[(long)(n0 + sr) * K + k0 + sh];
#if HAVE_ASYNC_LDS
            __builtin_amdgcn_global_load_async_to_lds_b128(
                (__attribute__((address_space(1))) v4i*)ga,
                (__attribute__((address_space(3))) v4i*)&As[sr][sh], 0, 0);
            __builtin_amdgcn_global_load_async_to_lds_b128(
                (__attribute__((address_space(1))) v4i*)(ga + 8),
                (__attribute__((address_space(3))) v4i*)&As[sr][sh + 8], 0, 0);
            __builtin_amdgcn_global_load_async_to_lds_b128(
                (__attribute__((address_space(1))) v4i*)gb,
                (__attribute__((address_space(3))) v4i*)&BsT[sr][sh], 0, 0);
            __builtin_amdgcn_global_load_async_to_lds_b128(
                (__attribute__((address_space(1))) v4i*)(gb + 8),
                (__attribute__((address_space(3))) v4i*)&BsT[sr][sh + 8], 0, 0);
            __builtin_amdgcn_s_wait_asynccnt(0);
#else
            const uint4* ga4 = (const uint4*)ga;
            const uint4* gb4 = (const uint4*)gb;
            *(uint4*)&As [sr][sh]     = ga4[0];
            *(uint4*)&As [sr][sh + 8] = ga4[1];
            *(uint4*)&BsT[sr][sh]     = gb4[0];
            *(uint4*)&BsT[sr][sh + 8] = gb4[1];
#endif
        } else {
            for (int i = tid; i < TM * TK; i += 128) {
                int r = i >> 5, kk = i & 31;
                int gr = m0 + r, gk = k0 + kk;
                As[r][kk] = (gr < M && gk < K) ? A[(long)gr * K + gk] : (unsigned short)0;
            }
            for (int i = tid; i < TN * TK; i += 128) {
                int cc = i >> 5, kk = i & 31;
                int gc = n0 + cc, gk = k0 + kk;
                BsT[cc][kk] = (gc < N && gk < K) ? Bt[(long)gc * K + gk] : (unsigned short)0;
            }
        }
        __syncthreads();

        FragU af[2], bfr[2];
        #pragma unroll
        for (int ri = 0; ri < 2; ++ri) {
            int row = wr + ri * 16 + mA;
            af[ri].q[0] = *(const uint4*)&As[row][kSelA];        // K = kSelA..kSelA+7
            af[ri].q[1] = *(const uint4*)&As[row][kSelA + 16];   // K = kSelA+16..kSelA+23
        }
        #pragma unroll
        for (int ci = 0; ci < 2; ++ci) {
            int col = wc + ci * 16 + nBl;
            bfr[ci].q[0] = *(const uint4*)&BsT[col][kSelB];      // K = kSelB..kSelB+7
            bfr[ci].q[1] = *(const uint4*)&BsT[col][kSelB + 16]; // K = kSelB+16..kSelB+23
        }
        #pragma unroll
        for (int ri = 0; ri < 2; ++ri)
        #pragma unroll
        for (int ci = 0; ci < 2; ++ci) {
            acc[ri][ci] = __builtin_amdgcn_wmma_f32_16x16x32_bf16(
                false, af[ri].v, false, bfr[ci].v, (short)0, acc[ri][ci], false, false);
        }
        __syncthreads();
    }

    // epilogue: C/D layout -> VGPR i: lanes 0-15 = row i, lanes 16-31 = row i+8
    int lm = lane & 15;
    int rshift = (lane < 16) ? 0 : 8;
    #pragma unroll
    for (int ri = 0; ri < 2; ++ri)
    #pragma unroll
    for (int ci = 0; ci < 2; ++ci) {
        #pragma unroll
        for (int i = 0; i < 8; ++i) {
            int row = m0 + wr + ri * 16 + i + rshift;
            int col = n0 + wc + ci * 16 + lm;
            if (row < M && col < N) {
                float v = acc[ri][ci][i];
                if (flags & 4) v += rowbias[(row / rows_per_batch) * rb_ld + col];
                if (flags & 1) v = lrelu(v * scale[col] + bias[col]);
                if (flags & 2) Cbf[cOff + (long)row * N + col] = f2bf(v);
                else           Cout[cOff + (long)row * N + col] = v;
            }
        }
    }
}

// ---------------------------------------------------------------------------
// Top-k(20) of pd[n,m] = 2*G[n,m] - xx[m] (the -xx[n] term is row-constant).
// One wave per point; scores staged in LDS; 20 rounds of wave argmax.
// ---------------------------------------------------------------------------
__global__ __launch_bounds__(256) void k_topk(const float* __restrict__ G,
                                              const float* __restrict__ xx,
                                              int* __restrict__ idx, int N)
{
    __shared__ float srow[8][NPTS];
    int wave = threadIdx.x >> 5, lane = threadIdx.x & 31;
    int p = blockIdx.x * 8 + wave;               // point id in [0, B*N)
    if (p >= NB * N) return;
    int b = p / N;
    const float* g  = G  + (long)p * N;
    const float* xb = xx + (long)b * N;
    float* row = srow[wave];
    for (int m = lane; m < N; m += 32) row[m] = 2.f * g[m] - xb[m];

    for (int j = 0; j < KNN; ++j) {
        float best = -INFINITY;
        int   bi   = N;
        for (int m = lane; m < N; m += 32) {
            float v = row[m];
            if (v > best) { best = v; bi = m; }
        }
        #pragma unroll
        for (int off = 16; off > 0; off >>= 1) {
            float ov = __shfl_xor(best, off, 32);
            int   oi = __shfl_xor(bi,   off, 32);
            if (ov > best || (ov == best && oi < bi)) { best = ov; bi = oi; }
        }
        if ((bi & 31) == lane && bi < N) row[bi] = -INFINITY;   // owner lane clears
        if (lane == 0) idx[(long)p * KNN + j] = bi;
    }
}

// ---------------------------------------------------------------------------
// Fused Local-Attention-Block: one block per point.
//   q = lrelu((wq@x_n)*sq+bq)            [C3]
//   v_j = lrelu((wv@fea_j)*sv+bv)        [C3]  fea built on the fly
//   s_j = lrelu((ws.(q+v_j))*ss+bs) -> softmax over j -> out = sum_j v_j*sc_j
// ---------------------------------------------------------------------------
__global__ __launch_bounds__(256) void k_lab(
    const float* __restrict__ xin, const int* __restrict__ idx,
    const float* __restrict__ wq, const float* __restrict__ wv, const float* __restrict__ wss,
    const float* __restrict__ sq, const float* __restrict__ bq,
    const float* __restrict__ sv, const float* __restrict__ bv,
    const float* __restrict__ ssp, const float* __restrict__ bsp,
    float* __restrict__ xout,
    int Cin, int Cf, int C3, int N, int first)
{
    __shared__ float xc[128];
    __shared__ float nbr[KNN * 128];
    __shared__ float qv[256];
    __shared__ float vv[KNN * 256];
    __shared__ float d2s[KNN];
    __shared__ float sj[KNN];

    int b = blockIdx.x / N, n = blockIdx.x % N;
    int tid = threadIdx.x;
    const long xoff = (long)b * Cin * N;

    for (int c = tid; c < Cin; c += 256) xc[c] = xin[xoff + (long)c * N + n];
    const int* id = idx + (long)(b * N + n) * KNN;
    for (int i = tid; i < KNN * Cin; i += 256) {
        int j = i / Cin, c = i % Cin;
        nbr[j * Cin + c] = xin[xoff + (long)c * N + id[j]];
    }
    __syncthreads();

    if (first && tid < KNN) {
        float a = 0.f;
        for (int c = 0; c < Cin; ++c) { float d = nbr[tid * Cin + c] - xc[c]; a += d * d; }
        d2s[tid] = a;
    }
    for (int o = tid; o < C3; o += 256) {
        float a = 0.f;
        for (int c = 0; c < Cin; ++c) a += wq[(long)o * Cin + c] * xc[c];
        qv[o] = lrelu(a * sq[o] + bq[o]);
    }
    __syncthreads();

    for (int i = tid; i < KNN * C3; i += 256) {
        int j = i / C3, o = i % C3;
        const float* w = wv + (long)o * Cf;
        float a = 0.f;
        if (first) {
            a = w[0] * d2s[j];                          // fea = [d2, diff, center, nbr]
            for (int c = 0; c < Cin; ++c) {
                float nb = nbr[j * Cin + c], ce = xc[c];
                a += w[1 + c] * (nb - ce) + w[1 + Cin + c] * ce + w[1 + 2 * Cin + c] * nb;
            }
        } else {                                        // fea = [diff, center]
            for (int c = 0; c < Cin; ++c) {
                float nb = nbr[j * Cin + c], ce = xc[c];
                a += w[c] * (nb - ce) + w[Cin + c] * ce;
            }
        }
        vv[j * C3 + o] = lrelu(a * sv[o] + bv[o]);
    }
    __syncthreads();

    if (tid < KNN) {
        float a = 0.f;
        for (int o = 0; o < C3; ++o) a += wss[o] * (qv[o] + vv[tid * C3 + o]);
        sj[tid] = lrelu(a * ssp[0] + bsp[0]);
    }
    __syncthreads();

    if (tid == 0) {
        float mx = -INFINITY;
        for (int j = 0; j < KNN; ++j) mx = fmaxf(mx, sj[j]);
        float sm = 0.f;
        for (int j = 0; j < KNN; ++j) { float e = __expf(sj[j] - mx); sj[j] = e; sm += e; }
        float inv = 1.f / sm;
        for (int j = 0; j < KNN; ++j) sj[j] *= inv;
    }
    __syncthreads();

    for (int o = tid; o < C3; o += 256) {
        float a = 0.f;
        for (int j = 0; j < KNN; ++j) a += vv[j * C3 + o] * sj[j];
        xout[(long)b * C3 * N + (long)o * N + n] = a;
    }
}

// concat x1..x4 channel-wise, transpose to [B*N, 512] bf16
__global__ void k_featT(const float* __restrict__ x1, const float* __restrict__ x2,
                        const float* __restrict__ x3, const float* __restrict__ x4,
                        unsigned short* __restrict__ out, int N)
{
    long i = (long)blockIdx.x * blockDim.x + threadIdx.x;
    long total = (long)NB * N * 512;
    if (i >= total) return;
    int c  = (int)(i % 512);
    long bn = i / 512;
    int b = (int)(bn / N), n = (int)(bn % N);
    float v;
    if      (c < 64)  v = x1[((long)b * 64  + c)        * N + n];
    else if (c < 128) v = x2[((long)b * 64  + (c - 64)) * N + n];
    else if (c < 256) v = x3[((long)b * 128 + (c - 128)) * N + n];
    else              v = x4[((long)b * 256 + (c - 256)) * N + n];
    out[i] = f2bf(v);
}

// Generic small linear: out[b,o] = post( sum_c in[b,c]*w[o,c] )
// flags: 1 = add ab[o] ; 2 = *= sc[o] += bi[o] ; 4 = lrelu
__global__ void k_lin(const float* __restrict__ in, const float* __restrict__ w,
                      const float* __restrict__ ab, const float* __restrict__ sc,
                      const float* __restrict__ bi, float* __restrict__ out,
                      int Bt, int K, int O, int flags)
{
    int i = blockIdx.x * blockDim.x + threadIdx.x;
    if (i >= Bt * O) return;
    int b = i / O, o = i % O;
    const float* ib = in + (long)b * K;
    const float* wo = w  + (long)o * K;
    float acc = 0.f;
    for (int c = 0; c < K; ++c) acc += ib[c] * wo[c];
    if (flags & 1) acc += ab[o];
    if (flags & 2) acc = acc * sc[o] + bi[o];
    if (flags & 4) acc = lrelu(acc);
    out[i] = acc;
}

// s[b,n] = lrelu( (sB[b,n,:].w4) * s4 + b4 )
__global__ void k_sfin(const unsigned short* __restrict__ sB, const float* __restrict__ w4,
                       const float* __restrict__ s4, const float* __restrict__ b4,
                       float* __restrict__ sout, int Bn)
{
    int i = blockIdx.x * blockDim.x + threadIdx.x;
    if (i >= Bn) return;
    const unsigned short* r = sB + (long)i * 64;
    float acc = 0.f;
    for (int c = 0; c < 64; ++c) acc += bf2f(r[c]) * w4[c];
    sout[i] = lrelu(acc * s4[0] + b4[0]);
}

// per-batch softmax over N points
__global__ __launch_bounds__(256) void k_softmax(const float* __restrict__ s,
                                                 float* __restrict__ sc, int N)
{
    __shared__ float red[256];
    int b = blockIdx.x, t = threadIdx.x;
    const float* sb = s + (long)b * N;
    float* scb = sc + (long)b * N;
    float mx = -INFINITY;
    for (int n = t; n < N; n += 256) mx = fmaxf(mx, sb[n]);
    red[t] = mx; __syncthreads();
    for (int o = 128; o; o >>= 1) { if (t < o) red[t] = fmaxf(red[t], red[t + o]); __syncthreads(); }
    mx = red[0]; __syncthreads();
    float sum = 0.f;
    for (int n = t; n < N; n += 256) { float e = __expf(sb[n] - mx); scb[n] = e; sum += e; }
    red[t] = sum; __syncthreads();
    for (int o = 128; o; o >>= 1) { if (t < o) red[t] += red[t + o]; __syncthreads(); }
    float inv = 1.f / red[0];
    for (int n = t; n < N; n += 256) scb[n] *= inv;
}

// g[b,c] = sum_n values_bf[b,n,c] * sc[b,n]
__global__ void k_wsum(const unsigned short* __restrict__ vals, const float* __restrict__ sc,
                       float* __restrict__ g, int N, int Cc)
{
    int c = blockIdx.x * blockDim.x + threadIdx.x;
    int b = blockIdx.y;
    if (c >= Cc) return;
    const unsigned short* vb = vals + (long)b * N * Cc + c;
    const float* scb = sc + (long)b * N;
    float acc = 0.f;
    for (int n = 0; n < N; ++n) acc += bf2f(vb[(long)n * Cc]) * scb[n];
    g[(long)b * Cc + c] = acc;
}

// ---------------------------------------------------------------------------
// Host-side orchestration
// ---------------------------------------------------------------------------
static inline int nblk(long n, int t) { return (int)((n + t - 1) / t); }

extern "C" void kernel_launch(void* const* d_in, const int* in_sizes, int n_in,
                              void* d_out, int out_size, void* d_ws, size_t ws_size,
                              hipStream_t stream)
{
    (void)in_sizes; (void)n_in; (void)out_size; (void)ws_size;
    const int N = NPTS;

    const float* X = (const float*)d_in[0];
    // input index map (dict flattening order)
    const int L1 = 2, L2 = 11, L3 = 20, L4 = 29, GA = 38, HD = 50;
    // lab offsets: wq+0 wv+1 ws+2 sq+3 bq+4 sv+5 bv+6 ss+7 bs+8
    // gab: w1 s1 b1 w2 s2 b2 w3 s3 b3 w4 s4 b4
    // head: w1 s6 b6 w2 bias2 s7 b7 w3 bias3

    // ---- workspace arena --------------------------------------------------
    size_t off = 0;
    auto alloc = [&](size_t bytes) -> char* {
        char* p = (char*)d_ws + off;
        off = (off + bytes + 255) & ~(size_t)255;
        return p;
    };
    float*          G        = (float*)alloc((size_t)NB * N * N * 4);          // 64 MB
    float*          xxb      = (float*)alloc((size_t)NB * N * 4);
    int*            idxb     = (int*)alloc((size_t)NB * N * KNN * 4);
    unsigned short* gA       = (unsigned short*)alloc((size_t)NB * N * 128 * 2);
    float*          x1       = (float*)alloc((size_t)NB * 64  * N * 4);
    float*          x2       = (float*)alloc((size_t)NB * 64  * N * 4);
    float*          x3       = (float*)alloc((size_t)NB * 128 * N * 4);
    float*          x4       = (float*)alloc((size_t)NB * 256 * N * 4);
    unsigned short* featT    = (unsigned short*)alloc((size_t)NB * N * 512 * 2);
    float*          xglo     = (float*)alloc((size_t)NB * 2 * N * 4);
    unsigned short* w1b      = (unsigned short*)alloc((size_t)2048 * 512 * 2);
    unsigned short* w2b      = (unsigned short*)alloc((size_t)512 * 2048 * 2);
    unsigned short* w3b      = (unsigned short*)alloc((size_t)64 * 512 * 2);
    unsigned short* valuesbf = (unsigned short*)alloc((size_t)NB * N * 2048 * 2); // 64 MB
    float*          xw2      = (float*)alloc((size_t)NB * 512 * 4);
    unsigned short* sA       = (unsigned short*)alloc((size_t)NB * N * 512 * 2);
    unsigned short* sB       = (unsigned short*)alloc((size_t)NB * N * 64 * 2);
    float*          sraw     = (float*)alloc((size_t)NB * N * 4);
    float*          scb      = (float*)alloc((size_t)NB * N * 4);
    float*          gvec     = (float*)alloc((size_t)NB * 2048 * 4);
    float*          h1       = (float*)alloc((size_t)NB * 512 * 4);
    float*          h2       = (float*)alloc((size_t)NB * 256 * 4);

    // ---- one graph-feature + LAB stage ------------------------------------
    auto stage = [&](const float* xin, int Cin, int C3, int first, int pb, float* xout) {
        long tot = (long)NB * Cin * N;
        k_chw2nc<<<nblk(tot, 256), 256, 0, stream>>>(xin, gA, Cin, N);
        k_xx    <<<nblk((long)NB * N, 256), 256, 0, stream>>>(xin, xxb, Cin, N);
        {   // Gram: per-batch  G[b] = (x^T)[N,Cin] * ((x^T)[N,Cin])^T  (A == Bt)
            dim3 grid(N / TN, N / TM, NB);
            k_gemm_bf16<<<grid, 128, 0, stream>>>(gA, gA, G, nullptr,
                N, N, Cin, (long)N * Cin, (long)N * Cin, (long)N * N,
                nullptr, nullptr, nullptr, 1, 0, 0);
        }
        k_topk<<<nblk((long)NB * N, 8) , 256, 0, stream>>>(G, xxb, idxb, N);
        int Cf = first ? (3 * Cin + 1) : (2 * Cin);
        k_lab<<<NB * N, 256, 0, stream>>>(xin, idxb,
            (const float*)d_in[pb + 0], (const float*)d_in[pb + 1], (const float*)d_in[pb + 2],
            (const float*)d_in[pb + 3], (const float*)d_in[pb + 4],
            (const float*)d_in[pb + 5], (const float*)d_in[pb + 6],
            (const float*)d_in[pb + 7], (const float*)d_in[pb + 8],
            xout, Cin, Cf, C3, N, first);
    };

    stage(X,  3,   64,  1, L1, x1);
    stage(x1, 64,  64,  0, L2, x2);
    stage(x2, 64,  128, 0, L3, x3);
    stage(x3, 128, 256, 0, L4, x4);

    // ---- GAB ---------------------------------------------------------------
    k_featT<<<nblk((long)NB * N * 512, 256), 256, 0, stream>>>(x1, x2, x3, x4, featT, N);
    k_xglo <<<nblk((long)NB * N, 256), 256, 0, stream>>>(X, xglo, N);

    // values = lrelu((feat @ w1^T)*s1+b1)  -> bf16 [B*N, 2048]
    k_tobf<<<nblk((long)2048 * 512, 256), 256, 0, stream>>>((const float*)d_in[GA + 0], w1b, (long)2048 * 512);
    {
        dim3 grid(2048 / TN, (NB * N) / TM, 1);
        k_gemm_bf16<<<grid, 128, 0, stream>>>(featT, w1b, nullptr, valuesbf,
            NB * N, 2048, 512, 0, 0, 0,
            (const float*)d_in[GA + 1], (const float*)d_in[GA + 2],
            nullptr, 1, 0, 1 | 2);
    }
    // xw2[b,:] = xglo[b] @ w2^T   (folds the +x_glo broadcast into a row bias)
    k_lin<<<nblk((long)NB * 512, 256), 256, 0, stream>>>(xglo, (const float*)d_in[GA + 3],
        nullptr, nullptr, nullptr, xw2, NB, 2048, 512, 0);
    // sA = lrelu(((values @ w2^T) + xw2[b])*s2+b2) -> bf16 [B*N,512]
    k_tobf<<<nblk((long)512 * 2048, 256), 256, 0, stream>>>((const float*)d_in[GA + 3], w2b, (long)512 * 2048);
    {
        dim3 grid(512 / TN, (NB * N) / TM, 1);
        k_gemm_bf16<<<grid, 128, 0, stream>>>(valuesbf, w2b, nullptr, sA,
            NB * N, 512, 2048, 0, 0, 0,
            (const float*)d_in[GA + 4], (const float*)d_in[GA + 5],
            xw2, N, 512, 1 | 2 | 4);
    }
    // sB = lrelu((sA @ w3^T)*s3+b3) -> bf16 [B*N,64]
    k_tobf<<<nblk((long)64 * 512, 256), 256, 0, stream>>>((const float*)d_in[GA + 6], w3b, (long)64 * 512);
    {
        dim3 grid(64 / TN, (NB * N) / TM, 1);
        k_gemm_bf16<<<grid, 128, 0, stream>>>(sA, w3b, nullptr, sB,
            NB * N, 64, 512, 0, 0, 0,
            (const float*)d_in[GA + 7], (const float*)d_in[GA + 8],
            nullptr, 1, 0, 1 | 2);
    }
    // s = lrelu((sB . w4)*s4+b4); softmax over points; g = sum_n values*sc
    k_sfin<<<nblk((long)NB * N, 256), 256, 0, stream>>>(sB, (const float*)d_in[GA + 9],
        (const float*)d_in[GA + 10], (const float*)d_in[GA + 11], sraw, NB * N);
    k_softmax<<<NB, 256, 0, stream>>>(sraw, scb, N);
    {
        dim3 grid(2048 / 256, NB, 1);
        k_wsum<<<grid, 256, 0, stream>>>(valuesbf, scb, gvec, N, 2048);
    }

    // ---- head --------------------------------------------------------------
    k_lin<<<nblk((long)NB * 512, 256), 256, 0, stream>>>(gvec, (const float*)d_in[HD + 0],
        nullptr, (const float*)d_in[HD + 1], (const float*)d_in[HD + 2], h1,
        NB, 2048, 512, 2 | 4);
    k_lin<<<nblk((long)NB * 256, 256), 256, 0, stream>>>(h1, (const float*)d_in[HD + 3],
        (const float*)d_in[HD + 4], (const float*)d_in[HD + 5], (const float*)d_in[HD + 6], h2,
        NB, 512, 256, 1 | 2 | 4);
    k_lin<<<nblk((long)NB * 40, 256), 256, 0, stream>>>(h2, (const float*)d_in[HD + 7],
        (const float*)d_in[HD + 8], nullptr, nullptr, (float*)d_out,
        NB, 256, 40, 1);
}
